// MP2C_2903397892252
// MI455X (gfx1250) — matrix operand order repaired and checked by reading, not compiled, not run on hardware
//
#include <hip/hip_runtime.h>
#include <hip/hip_bf16.h>
#include <math.h>

#define BATCH 16
#define NPTS  8192
#define MROWS (BATCH * NPTS)   // 131072

typedef __attribute__((ext_vector_type(16))) __bf16   v16bf;
typedef __attribute__((ext_vector_type(8)))  float    v8f;
typedef __attribute__((ext_vector_type(4)))  unsigned u32x4;

union Frag { v16bf v; u32x4 q[2]; };

__device__ inline v8f zero8() { v8f z = {0.f,0.f,0.f,0.f,0.f,0.f,0.f,0.f}; return z; }

// A fragment (16x32 bf16, MxK): lane<16 -> row m=lane, K k0..k0+15
//                               lane>=16 -> row m=lane-16, K k0+16..k0+31
__device__ inline void load_a(Frag& f, const __bf16* rowk, int lane) {
  int o = (lane & 16) ? 16 : 0;
  f.q[0] = *(const u32x4*)(rowk + o);
  f.q[1] = *(const u32x4*)(rowk + o + 8);
}
// B fragment (32x16 bf16, KxN): lane<16 -> col n=lane, K k0..k0+15; lane>=16 -> col n=lane-16, K k0+16..k0+31
__device__ inline void load_b(Frag& f, const __bf16* colk, int lane) {
  int o = (lane & 16) ? 16 : 0;
  f.q[0] = *(const u32x4*)(colk + o);
  f.q[1] = *(const u32x4*)(colk + o + 8);
}

// Order-preserving float<->uint encoding for atomicMax on signed floats
__device__ inline unsigned fenc(float x) {
  unsigned u = __float_as_uint(x);
  return (u & 0x80000000u) ? ~u : (u | 0x80000000u);
}
__device__ inline float fdec(unsigned e) {
  unsigned u = (e & 0x80000000u) ? (e ^ 0x80000000u) : ~e;
  return __uint_as_float(u);
}

// ---------------- init / convert / BN helpers ----------------

__global__ void init_kernel(float* stats, unsigned* gmax) {
  int i = blockIdx.x * 256 + threadIdx.x;
  if (i < 1280) stats[i] = 0.f;
  if (i < 16384) gmax[i] = 0x007FFFFFu;  // fenc(-inf)
}

__global__ void cvt_kernel(const float* __restrict__ src, __bf16* __restrict__ dst, int n) {
  int i = blockIdx.x * 256 + threadIdx.x;
  if (i < n) dst[i] = (__bf16)src[i];
}

// f1 = partial @ W1^T + b1 ; accumulate per-channel sum / sumsq (128 channels)
__global__ __launch_bounds__(128) void stats1_kernel(const float* __restrict__ P,
    const float* __restrict__ W1, const float* __restrict__ b1,
    float* __restrict__ s, float* __restrict__ sq) {
  int c = threadIdx.x;
  size_t row0 = (size_t)blockIdx.x * 1024;
  float w0 = W1[c*3], w1 = W1[c*3+1], w2 = W1[c*3+2], bb = b1[c];
  float a0 = 0.f, a1 = 0.f;
  for (int i = 0; i < 1024; ++i) {
    size_t r = row0 + i;
    float v = fmaf(w0, P[r*3], fmaf(w1, P[r*3+1], fmaf(w2, P[r*3+2], bb)));
    a0 += v; a1 += v * v;
  }
  atomicAdd(&s[c], a0);
  atomicAdd(&sq[c], a1);
}

__global__ void fin_bn_kernel(const float* __restrict__ s, const float* __restrict__ sq,
    const float* __restrict__ g, const float* __restrict__ be,
    float* __restrict__ A, float* __restrict__ Bc, float inv) {
  int c = threadIdx.x;
  float mu = s[c] * inv;
  float var = sq[c] * inv - mu * mu;
  float sc = g[c] * rsqrtf(var + 1e-5f);
  A[c] = sc;
  Bc[c] = be[c] - mu * sc;
}

// x1 = bf16(relu(bn(partial @ W1^T + b1)))   [M,128]
__global__ __launch_bounds__(128) void x1_kernel(const float* __restrict__ P,
    const float* __restrict__ W1, const float* __restrict__ b1,
    const float* __restrict__ A, const float* __restrict__ Bc, __bf16* __restrict__ X1) {
  int c = threadIdx.x;
  float w0 = W1[c*3], w1 = W1[c*3+1], w2 = W1[c*3+2], bb = b1[c];
  float sc = A[c], sh = Bc[c];
  size_t row0 = (size_t)blockIdx.x * 256;
  for (int i = 0; i < 256; ++i) {
    size_t r = row0 + i;
    float v = fmaf(w0, P[r*3], fmaf(w1, P[r*3+1], fmaf(w2, P[r*3+2], bb)));
    v = fmaxf(fmaf(v, sc, sh), 0.f);
    X1[r*128 + c] = (__bf16)v;
  }
}

// ---------------- WMMA GEMMs ----------------
// Wave tile 16x64 (4 accumulators), block = 8 waves = 32 rows x 256 cols.

// GEMM2: X1[M,128] @ W2bf[256,128]^T + b2 -> x2 bf16 [M,256]
__global__ __launch_bounds__(256) void gemm2_wmma(const __bf16* __restrict__ X,
    const __bf16* __restrict__ W, const float* __restrict__ bias, __bf16* __restrict__ Y) {
  const int K = 128, NT = 256;
  int lane = threadIdx.x & 31, wave = threadIdx.x >> 5;
  int mi = wave & 1, ni = wave >> 1;
  size_t row0 = (size_t)blockIdx.x * 32 + mi * 16;
  int col0 = ni * 64;
  size_t r = row0 + (lane & 15);
  v8f acc[4];
  for (int j = 0; j < 4; ++j) acc[j] = zero8();
  for (int k0 = 0; k0 < K; k0 += 32) {
    Frag a; load_a(a, X + r * K + k0, lane);
#pragma unroll
    for (int j = 0; j < 4; ++j) {
      int c = col0 + j * 16 + (lane & 15);
      Frag bf; load_b(bf, W + (size_t)c * K + k0, lane);
      acc[j] = __builtin_amdgcn_wmma_f32_16x16x32_bf16(false, a.v, false, bf.v,
                                                       (short)0, acc[j], false, false);
    }
  }
  int rh = (lane >> 4) * 8;
#pragma unroll
  for (int j = 0; j < 4; ++j) {
    int c = col0 + j * 16 + (lane & 15);
    float bc = bias[c];
#pragma unroll
    for (int v = 0; v < 8; ++v)
      Y[(row0 + rh + v) * NT + c] = (__bf16)(acc[j][v] + bc);
  }
}

// fg[b][c] = max_n x2[b,n,c]  -> bf16 [16,256]
__global__ __launch_bounds__(256) void maxfg_kernel(const __bf16* __restrict__ X2,
                                                    __bf16* __restrict__ FG) {
  int c = threadIdx.x;
  int b = blockIdx.x;
  size_t base = (size_t)b * NPTS;
  float m = -INFINITY;
  for (int n = 0; n < NPTS; ++n) m = fmaxf(m, (float)X2[(base + n) * 256 + c]);
  FG[b * 256 + c] = (__bf16)m;
}

// GEMM3: concat(fg, x2)[M,512] @ W3bf[512,512]^T + b3 -> f3 bf16 [M,512], fused sum/sumsq stats
__global__ __launch_bounds__(256) void gemm3_wmma(const __bf16* __restrict__ FG,
    const __bf16* __restrict__ X2, const __bf16* __restrict__ W,
    const float* __restrict__ bias, __bf16* __restrict__ Y,
    float* __restrict__ ssum, float* __restrict__ ssq) {
  const int K = 512, NT = 512;
  int lane = threadIdx.x & 31, wave = threadIdx.x >> 5;
  int mi = wave & 1, ni = wave >> 1;
  size_t row0 = (size_t)blockIdx.x * 32 + mi * 16;
  int col0 = blockIdx.y * 256 + ni * 64;
  size_t r = row0 + (lane & 15);
  size_t b = r >> 13;  // 8192 points per batch
  v8f acc[4];
  for (int j = 0; j < 4; ++j) acc[j] = zero8();
  for (int k0 = 0; k0 < K; k0 += 32) {
    const __bf16* arow = (k0 < 256) ? (FG + b * 256 + k0)
                                    : (X2 + r * 256 + (size_t)(k0 - 256));
    Frag a; load_a(a, arow, lane);
#pragma unroll
    for (int j = 0; j < 4; ++j) {
      int c = col0 + j * 16 + (lane & 15);
      Frag bf; load_b(bf, W + (size_t)c * K + k0, lane);
      acc[j] = __builtin_amdgcn_wmma_f32_16x16x32_bf16(false, a.v, false, bf.v,
                                                       (short)0, acc[j], false, false);
    }
  }
  int rh = (lane >> 4) * 8;
#pragma unroll
  for (int j = 0; j < 4; ++j) {
    int c = col0 + j * 16 + (lane & 15);
    float bc = bias[c];
    float s0 = 0.f, s1 = 0.f;
#pragma unroll
    for (int v = 0; v < 8; ++v) {
      float t = acc[j][v] + bc;
      s0 += t; s1 += t * t;
      Y[(row0 + rh + v) * NT + c] = (__bf16)t;
    }
    s0 += __shfl_xor(s0, 16, 32);
    s1 += __shfl_xor(s1, 16, 32);
    if (lane < 16) {
      atomicAdd(&ssum[c], s0);
      atomicAdd(&ssq[c], s1);
    }
  }
}

// in-place: x4 = bf16(relu(bn2(f3)))
__global__ __launch_bounds__(256) void x4_kernel(__bf16* __restrict__ F3,
    const float* __restrict__ A, const float* __restrict__ Bc) {
  size_t idx = (size_t)blockIdx.x * 256 + threadIdx.x;
  int c = (int)(idx & 511);
  float v = (float)F3[idx];
  v = fmaxf(fmaf(v, A[c], Bc[c]), 0.f);
  F3[idx] = (__bf16)v;
}

// ---- GEMM4: x4[M,512] @ W4bf[1024,512]^T + b4, fused max-over-N via atomicMax.
// B tile (256 cols x 32 K = 16KB) staged in LDS with GLOBAL_LOAD_ASYNC_TO_LDS_B128,
// double-buffered (ASYNCcnt + s_wait_asynccnt + barrier). Rows padded to 80B so the
// 16-lane ds_load_b128 fragment reads hit distinct banks.
#define BROW 80
#define BBUF (256 * BROW)

// each thread copies one 64B weight row as 4 async b128 ops; INST_OFFSET applies to
// both the global and the LDS address (ISA 15.18.3), so one base pair suffices.
__device__ inline void issue_b_tile(const __bf16* gsrc, unsigned ldsrow) {
  asm volatile(
      "global_load_async_to_lds_b128 %0, %1, off\n\t"
      "global_load_async_to_lds_b128 %0, %1, off offset:16\n\t"
      "global_load_async_to_lds_b128 %0, %1, off offset:32\n\t"
      "global_load_async_to_lds_b128 %0, %1, off offset:48"
      :: "v"(ldsrow), "v"((unsigned long long)(size_t)gsrc)
      : "memory");
}

__global__ __launch_bounds__(256) void gemm4_wmma(const __bf16* __restrict__ X,
    const __bf16* __restrict__ W, const float* __restrict__ bias,
    unsigned* __restrict__ gmax) {
  const int K = 512;
  __shared__ __align__(16) char lbuf[2][BBUF];
  int lane = threadIdx.x & 31, wave = threadIdx.x >> 5;
  int mi = wave & 1, ni = wave >> 1;
  size_t row0 = (size_t)blockIdx.x * 32 + mi * 16;
  int c0 = blockIdx.y * 256;                 // block's N window
  size_t r = row0 + (lane & 15);
  // generic pointer into __shared__: low 32 bits are the LDS byte offset
  unsigned lrow0 = (unsigned)(size_t)(&lbuf[0][threadIdx.x * BROW]);
  unsigned lrow1 = (unsigned)(size_t)(&lbuf[1][threadIdx.x * BROW]);
  const __bf16* wrow = W + (size_t)(c0 + threadIdx.x) * K;  // this thread's weight row

  v8f acc[4];
  for (int j = 0; j < 4; ++j) acc[j] = zero8();

  issue_b_tile(wrow, lrow0);  // prologue: K-chunk 0 -> buf0
  for (int i = 0; i < 16; ++i) {
    int k0 = i * 32;
    if (i < 15) {
      issue_b_tile(wrow + (k0 + 32), (i & 1) ? lrow0 : lrow1);
      asm volatile("s_wait_asynccnt 0x4" ::: "memory");  // current chunk landed
    } else {
      asm volatile("s_wait_asynccnt 0x0" ::: "memory");
    }
    __syncthreads();  // all waves' async writes for this chunk visible

    Frag a; load_a(a, X + r * K + k0, lane);
    if (k0 + 32 < K) __builtin_prefetch(X + r * K + k0 + 32, 0, 3);
    const char* lb = &lbuf[i & 1][0];
#pragma unroll
    for (int j = 0; j < 4; ++j) {
      int ct = ni * 64 + j * 16 + (lane & 15);             // tile-local col
      const char* bp = lb + ct * BROW + ((lane & 16) ? 32 : 0);
      Frag bf;
      bf.q[0] = *(const u32x4*)bp;
      bf.q[1] = *(const u32x4*)(bp + 16);
      acc[j] = __builtin_amdgcn_wmma_f32_16x16x32_bf16(false, a.v, false, bf.v,
                                                       (short)0, acc[j], false, false);
    }
    __syncthreads();  // everyone done reading before buffer is overwritten
  }

  int bb = (int)(row0 >> 13);
#pragma unroll
  for (int j = 0; j < 4; ++j) {
    int c = c0 + ni * 64 + j * 16 + (lane & 15);
    float bc = bias[c];
    float m = -INFINITY;
#pragma unroll
    for (int v = 0; v < 8; ++v) m = fmaxf(m, acc[j][v] + bc);
    m = fmaxf(m, __shfl_xor(m, 16, 32));
    if (lane < 16) atomicMax(&gmax[bb * 1024 + c], fenc(m));
  }
}

__global__ void featmax_kernel(const unsigned* __restrict__ gmax, float* __restrict__ feat) {
  int i = blockIdx.x * 256 + threadIdx.x;
  if (i >= 16384) return;
  int b = i >> 10, c = i & 1023;
  feat[(size_t)b * 2048 + c] = fdec(gmax[i]);
}

// ---------------- memory module + decoder ----------------

__global__ __launch_bounds__(256) void gates_kernel(
    const float* __restrict__ Wf, const float* __restrict__ bf_,
    const float* __restrict__ Wi, const float* __restrict__ bi_,
    const float* __restrict__ Wo, const float* __restrict__ bo_,
    const float* __restrict__ Wc, const float* __restrict__ bc_,
    const float* __restrict__ memp, float* __restrict__ feat) {
  int t = blockIdx.x * 256 + threadIdx.x;
  if (t >= BATCH * 1024) return;
  int b = t >> 10, j = t & 1023;
  const float* x = feat + (size_t)b * 2048;  // first half = feat_global
  const float* wf = Wf + (size_t)j * 1024;
  const float* wi = Wi + (size_t)j * 1024;
  const float* wo = Wo + (size_t)j * 1024;
  const float* wc = Wc + (size_t)j * 1024;
  float af = bf_[j], ai = bi_[j], ao = bo_[j], ac = bc_[j];
  for (int k = 0; k < 1024; ++k) {
    float xv = x[k];
    af = fmaf(wf[k], xv, af);
    ai = fmaf(wi[k], xv, ai);
    ao = fmaf(wo[k], xv, ao);
    ac = fmaf(wc[k], xv, ac);
  }
  float gf = 1.f / (1.f + __expf(-af));
  float gi = 1.f / (1.f + __expf(-ai));
  float go = 1.f / (1.f + __expf(-ao));
  float cd = tanhf(ac);
  float h = go * tanhf(gf * memp[j] + gi * cd);
  feat[(size_t)b * 2048 + 1024 + j] = h;
}

// Decoder layer: Y[16,No] = (relu?)(X[16,Kd] @ W[No,Kd]^T + bias). Block per output column,
// split-K over 256 threads (coalesced weight reads), shfl + LDS reduce.
__global__ __launch_bounds__(256) void fc_kernel(const float* __restrict__ X,
    const float* __restrict__ W, const float* __restrict__ bias,
    float* __restrict__ Y, int Kd, int No, int dorelu) {
  __shared__ float red[8][16];
  int o = blockIdx.x;
  int tid = threadIdx.x, lane = tid & 31, wv = tid >> 5;
  float acc[16];
#pragma unroll
  for (int b = 0; b < 16; ++b) acc[b] = 0.f;
  const float* wr = W + (size_t)o * Kd;
  for (int k = tid; k < Kd; k += 256) {
    float w = wr[k];
#pragma unroll
    for (int b = 0; b < 16; ++b) acc[b] = fmaf(w, X[b * Kd + k], acc[b]);
  }
#pragma unroll
  for (int b = 0; b < 16; ++b) {
    float v = acc[b];
    for (int off = 16; off; off >>= 1) v += __shfl_down(v, off, 32);
    if (lane == 0) red[wv][b] = v;
  }
  __syncthreads();
  if (tid < 16) {
    float v = 0.f;
#pragma unroll
    for (int w = 0; w < 8; ++w) v += red[w][tid];
    v += bias[o];
    if (dorelu) v = fmaxf(v, 0.f);
    Y[tid * No + o] = v;
  }
}

// ---------------- launch ----------------

extern "C" void kernel_launch(void* const* d_in, const int* in_sizes, int n_in,
                              void* d_out, int out_size, void* d_ws, size_t ws_size,
                              hipStream_t stream) {
  (void)in_sizes; (void)n_in; (void)out_size; (void)ws_size;
  const float* partial = (const float*)d_in[0];
  const float* W1  = (const float*)d_in[1];
  const float* b1  = (const float*)d_in[2];
  const float* g1  = (const float*)d_in[3];
  const float* be1 = (const float*)d_in[4];
  const float* W2  = (const float*)d_in[5];
  const float* b2  = (const float*)d_in[6];
  const float* W3  = (const float*)d_in[7];
  const float* b3  = (const float*)d_in[8];
  const float* g2  = (const float*)d_in[9];
  const float* be2 = (const float*)d_in[10];
  const float* W4  = (const float*)d_in[11];
  const float* b4  = (const float*)d_in[12];
  const float* Wf  = (const float*)d_in[13];
  const float* bf_ = (const float*)d_in[14];
  const float* Wi  = (const float*)d_in[15];
  const float* bi_ = (const float*)d_in[16];
  const float* Wo  = (const float*)d_in[17];
  const float* bo_ = (const float*)d_in[18];
  const float* Wc  = (const float*)d_in[19];
  const float* bc_ = (const float*)d_in[20];
  const float* memp= (const float*)d_in[21];
  const float* D1w = (const float*)d_in[22];
  const float* D1b = (const float*)d_in[23];
  const float* D2w = (const float*)d_in[24];
  const float* D2b = (const float*)d_in[25];
  const float* D3w = (const float*)d_in[26];
  const float* D3b = (const float*)d_in[27];
  const float* D4w = (const float*)d_in[28];
  const float* D4b = (const float*)d_in[29];

  char* w = (char*)d_ws;
  size_t off = 0;
  auto alloc = [&](size_t bytes) -> void* {
    void* p = w + off;
    off += (bytes + 255) & ~(size_t)255;
    return p;
  };
  __bf16* x1bf  = (__bf16*)alloc((size_t)MROWS * 128 * 2);  //  32 MB
  __bf16* x2bf  = (__bf16*)alloc((size_t)MROWS * 256 * 2);  //  64 MB
  __bf16* f3bf  = (__bf16*)alloc((size_t)MROWS * 512 * 2);  // 128 MB
  __bf16* w2bf  = (__bf16*)alloc(256 * 128 * 2);
  __bf16* w3bf  = (__bf16*)alloc(512 * 512 * 2);
  __bf16* w4bf  = (__bf16*)alloc(1024 * 512 * 2);
  __bf16* fgbf  = (__bf16*)alloc(16 * 256 * 2);
  float*  stats = (float*)alloc(1280 * 4);   // s1sum[128] s1sq[128] s3sum[512] s3sq[512]
  float*  bnA1  = (float*)alloc(128 * 4);
  float*  bnB1  = (float*)alloc(128 * 4);
  float*  bnA2  = (float*)alloc(512 * 4);
  float*  bnB2  = (float*)alloc(512 * 4);
  unsigned* gmax= (unsigned*)alloc(16384 * 4);
  float*  feat  = (float*)alloc(16 * 2048 * 4);
  float*  z1    = (float*)alloc(16 * 2048 * 4);
  float*  z2    = (float*)alloc(16 * 2048 * 4);
  float*  z3    = (float*)alloc(16 * 2048 * 4);
  float* s1sum = stats, *s1sq = stats + 128, *s3sum = stats + 256, *s3sq = stats + 768;

  init_kernel<<<64, 256, 0, stream>>>(stats, gmax);
  cvt_kernel<<<(32768 + 255) / 256, 256, 0, stream>>>(W2, w2bf, 32768);
  cvt_kernel<<<1024, 256, 0, stream>>>(W3, w3bf, 262144);
  cvt_kernel<<<2048, 256, 0, stream>>>(W4, w4bf, 524288);

  stats1_kernel<<<128, 128, 0, stream>>>(partial, W1, b1, s1sum, s1sq);
  fin_bn_kernel<<<1, 128, 0, stream>>>(s1sum, s1sq, g1, be1, bnA1, bnB1, 1.f / MROWS);
  x1_kernel<<<512, 128, 0, stream>>>(partial, W1, b1, bnA1, bnB1, x1bf);

  gemm2_wmma<<<dim3(MROWS / 32, 1), 256, 0, stream>>>(x1bf, w2bf, b2, x2bf);
  maxfg_kernel<<<16, 256, 0, stream>>>(x2bf, fgbf);

  gemm3_wmma<<<dim3(MROWS / 32, 2), 256, 0, stream>>>(fgbf, x2bf, w3bf, b3, f3bf, s3sum, s3sq);
  fin_bn_kernel<<<1, 512, 0, stream>>>(s3sum, s3sq, g2, be2, bnA2, bnB2, 1.f / MROWS);
  x4_kernel<<<(MROWS * 512) / 256, 256, 0, stream>>>(f3bf, bnA2, bnB2);

  gemm4_wmma<<<dim3(MROWS / 32, 4), 256, 0, stream>>>(f3bf, w4bf, b4, gmax);
  featmax_kernel<<<64, 256, 0, stream>>>(gmax, feat);

  gates_kernel<<<64, 256, 0, stream>>>(Wf, bf_, Wi, bi_, Wo, bo_, Wc, bc_, memp, feat);

  fc_kernel<<<2048, 256, 0, stream>>>(feat, D1w, D1b, z1, 2048, 2048, 1);
  fc_kernel<<<2048, 256, 0, stream>>>(z1, D2w, D2b, z2, 2048, 2048, 1);
  fc_kernel<<<2048, 256, 0, stream>>>(z2, D3w, D3b, z3, 2048, 2048, 1);
  fc_kernel<<<6144, 256, 0, stream>>>(z3, D4w, D4b, (float*)d_out, 2048, 6144, 0);
}
